// TorchJSDLoss_71159018160659
// MI455X (gfx1250) — matrix operand order repaired and checked by reading, not compiled, not run on hardware
//
#include <hip/hip_runtime.h>
#include <hip/hip_bf16.h>

// ---------------------------------------------------------------------------
// Fused JSD distillation loss for MI455X (gfx1250, wave32, WMMA bf16).
// Two-pass flash-softmax over V=32000, GEMMs via v_wmma_f32_16x16x32_bf16,
// double-buffered LDS staging (K-loop unrolled x2, static buffer offsets),
// statically specialized A/W staging (no divergent selects).
// ---------------------------------------------------------------------------

#define BT_      4096
#define VV_      32000
#define KS_      2048     // student hidden
#define KT_      4096     // teacher hidden
#define BM_      64
#define BN_      64
#define BK_      32
#define NTILES_  (VV_ / BN_)          // 500
#define NCHUNKS_ 20
#define TPC_     (NTILES_ / NCHUNKS_) // 25 tiles per pass-1 chunk
#define LDA_     (BK_ + 8)            // padded LDS row (bf16 elems), 80B stride
#define STAGE_   (BM_ * LDA_)         // one A (or W) stage buffer, ushort elems

typedef __attribute__((ext_vector_type(16))) __bf16 v16bf;
typedef __attribute__((ext_vector_type(8)))  __bf16 v8bf;
typedef __attribute__((ext_vector_type(2)))  __bf16 v2bf;
typedef __attribute__((ext_vector_type(8)))  float  v8f;

union V16U { v16bf v; v8bf h[2]; };

// Pack two f32 -> packed 2xbf16 (one dword).
#if __has_builtin(__builtin_amdgcn_cvt_pk_bf16_f32)
__device__ __forceinline__ unsigned pack_bf(float x, float y) {
    v2bf t = __builtin_amdgcn_cvt_pk_bf16_f32(x, y);
    return __builtin_bit_cast(unsigned, t);
}
#else
__device__ __forceinline__ unsigned pack_bf(float x, float y) {
    // round-half-up then one v_perm_b32 -> {y[31:16], x[31:16]}
    unsigned ux = __builtin_bit_cast(unsigned, x) + 0x8000u;
    unsigned uy = __builtin_bit_cast(unsigned, y) + 0x8000u;
    return __builtin_amdgcn_perm(uy, ux, 0x07060302u);
}
#endif

__device__ __forceinline__ v8bf lds_load8(const unsigned short* p) {
    uint4 t = *(const uint4*)p;               // 16B aligned by construction
    return __builtin_bit_cast(v8bf, t);
}

__device__ __forceinline__ v8f wmma_bf16(const V16U& a, const V16U& b, v8f c) {
    return __builtin_amdgcn_wmma_f32_16x16x32_bf16(
               false, a.v, false, b.v, (short)0, c, false, false);
}

// Stage loader: A rows 0..63 (st[0..1]) and W rows 0..63 (st[2..3]),
// statically split so base pointers are uniform (no exec-mask selects).
__device__ __forceinline__ void stage_load(
    const float* __restrict__ A, const float* __restrict__ W, int K,
    int row0, int col0, int kb, float4 (&st)[4], int tid)
{
    const int rowi = tid >> 3;            // 0..31
    const int c4   = (tid & 7) * 4;
    const float* pa0 = A + (size_t)(row0 + rowi)      * K + kb + c4;
    const float* pa1 = A + (size_t)(row0 + 32 + rowi) * K + kb + c4;
    const float* pw0 = W + (size_t)(col0 + rowi)      * K + kb + c4;
    const float* pw1 = W + (size_t)(col0 + 32 + rowi) * K + kb + c4;
    st[0] = *(const float4*)pa0;
    st[1] = *(const float4*)pa1;
    st[2] = *(const float4*)pw0;
    st[3] = *(const float4*)pw1;
    if (kb + 2 * BK_ < K) {               // warm caches two stages ahead
        __builtin_prefetch(pa0 + 2 * BK_, 0, 3);
        __builtin_prefetch(pa1 + 2 * BK_, 0, 3);
        __builtin_prefetch(pw0 + 2 * BK_, 0, 3);
        __builtin_prefetch(pw1 + 2 * BK_, 0, 3);
    }
}

__device__ __forceinline__ void stage_store(
    unsigned short* __restrict__ sA, unsigned short* __restrict__ sW,
    const float4 (&st)[4], int tid)
{
    const int rowi = tid >> 3;
    const int c4   = (tid & 7) * 4;
    unsigned* d0 = (unsigned*)(sA + rowi * LDA_ + c4);
    unsigned* d1 = (unsigned*)(sA + (32 + rowi) * LDA_ + c4);
    unsigned* d2 = (unsigned*)(sW + rowi * LDA_ + c4);
    unsigned* d3 = (unsigned*)(sW + (32 + rowi) * LDA_ + c4);
    d0[0] = pack_bf(st[0].x, st[0].y);  d0[1] = pack_bf(st[0].z, st[0].w);
    d1[0] = pack_bf(st[1].x, st[1].y);  d1[1] = pack_bf(st[1].z, st[1].w);
    d2[0] = pack_bf(st[2].x, st[2].y);  d2[1] = pack_bf(st[2].z, st[2].w);
    d3[0] = pack_bf(st[3].x, st[3].y);  d3[1] = pack_bf(st[3].z, st[3].w);
}

// Fragment assembly + 2 WMMAs (A and B share contiguous k = 16*hf + s per lane).
__device__ __forceinline__ void frag_mma(
    const unsigned short* __restrict__ sA, const unsigned short* __restrict__ sW,
    int mi, int ni, int hf, int l16, v8f& acc0, v8f& acc1)
{
    V16U a, b0, b1;
    const unsigned short* pa = sA + (mi + l16) * LDA_;
    a.h[0] = lds_load8(pa + hf * 16);
    a.h[1] = lds_load8(pa + hf * 16 + 8);
    const unsigned short* pb0 = sW + (ni + l16) * LDA_;
    b0.h[0] = lds_load8(pb0 + hf * 16);
    b0.h[1] = lds_load8(pb0 + hf * 16 + 8);
    const unsigned short* pb1 = sW + (ni + 16 + l16) * LDA_;
    b1.h[0] = lds_load8(pb1 + hf * 16);
    b1.h[1] = lds_load8(pb1 + hf * 16 + 8);
    acc0 = wmma_bf16(a, b0, acc0);
    acc1 = wmma_bf16(a, b1, acc1);
}

// 64x64 output block of A@W^T; each wave owns a 16x32 strip (2 wmma/K-step).
// Double-buffered, K-loop unrolled x2 so LDS buffer offsets are immediates.
// Requires K % (2*BK_) == 0  (holds: 2048, 4096).
__device__ __forceinline__ void gemm_dual(
    const float* __restrict__ A, const float* __restrict__ W, int K,
    int row0, int col0,
    unsigned short* __restrict__ sAbuf, unsigned short* __restrict__ sWbuf,
    v8f& acc0, v8f& acc1)
{
    const int tid  = threadIdx.x;
    const int lane = tid & 31;
    const int wv   = tid >> 5;
    const int mi   = (wv & 3) * 16;     // wave M offset
    const int ni   = (wv >> 2) * 32;    // wave N offset (two 16-col subtiles)
    const int hf   = lane >> 4;
    const int l16  = lane & 15;

    float4 st[4];
    stage_load(A, W, K, row0, col0, 0, st, tid);
    stage_store(sAbuf, sWbuf, st, tid);
    __syncthreads();

    for (int kb = 0; kb < K; kb += 2 * BK_) {
        // ---- step 0: compute from buf0; next block (kb+BK_) always exists
        stage_load(A, W, K, row0, col0, kb + BK_, st, tid);
        frag_mma(sAbuf, sWbuf, mi, ni, hf, l16, acc0, acc1);
        stage_store(sAbuf + STAGE_, sWbuf + STAGE_, st, tid);
        __syncthreads();
        // ---- step 1: compute from buf1; preload kb+2*BK_ if it exists
        const bool more = (kb + 2 * BK_) < K;
        if (more) stage_load(A, W, K, row0, col0, kb + 2 * BK_, st, tid);
        frag_mma(sAbuf + STAGE_, sWbuf + STAGE_, mi, ni, hf, l16, acc0, acc1);
        if (more) stage_store(sAbuf, sWbuf, st, tid);
        __syncthreads();
    }
}

// ---------------- Pass 1: per-row (max, sumexp) partials per chunk ---------
__global__ __launch_bounds__(256)
void pass1_kernel(const float* __restrict__ S, const float* __restrict__ T,
                  const float* __restrict__ Ws, const float* __restrict__ Wt,
                  float* __restrict__ stats /* [2][BT][NCHUNKS][2] */)
{
    __shared__ unsigned short sAbuf[2 * STAGE_];
    __shared__ unsigned short sWbuf[2 * STAGE_];
    __shared__ float cbuf[2][BM_][BN_];

    const int tid  = threadIdx.x;
    const int lane = tid & 31;
    const int wv   = tid >> 5;
    const int mi   = (wv & 3) * 16;
    const int ni   = (wv >> 2) * 32;
    const int hf   = lane >> 4;
    const int l16  = lane & 15;

    const int row0  = blockIdx.x * BM_;
    const int chunk = blockIdx.y;

    float rm = -__builtin_inff();   // running max  (threads 0..127)
    float rs = 0.0f;                // running sumexp

    for (int t = 0; t < TPC_; ++t) {
        const int col0 = (chunk * TPC_ + t) * BN_;
        v8f aS0 = {0.f,0.f,0.f,0.f,0.f,0.f,0.f,0.f}, aS1 = aS0;
        v8f aT0 = aS0, aT1 = aS0;
        gemm_dual(S, Ws, KS_, row0, col0, sAbuf, sWbuf, aS0, aS1);
        gemm_dual(T, Wt, KT_, row0, col0, sAbuf, sWbuf, aT0, aT1);

        #pragma unroll
        for (int i = 0; i < 8; ++i) {
            const int r = mi + i + hf * 8;
            cbuf[0][r][ni + l16]      = aS0[i];
            cbuf[0][r][ni + 16 + l16] = aS1[i];
            cbuf[1][r][ni + l16]      = aT0[i];
            cbuf[1][r][ni + 16 + l16] = aT1[i];
        }
        __syncthreads();

        if (tid < 128) {
            const int model = tid >> 6, r = tid & 63;
            float tm = -__builtin_inff();
            #pragma unroll 8
            for (int j = 0; j < BN_; ++j) tm = fmaxf(tm, cbuf[model][r][j]);
            const float nm = fmaxf(rm, tm);
            float s = 0.0f;
            #pragma unroll 8
            for (int j = 0; j < BN_; ++j) s += __expf(cbuf[model][r][j] - nm);
            rs = rs * __expf(rm - nm) + s;
            rm = nm;
        }
        __syncthreads();
    }

    if (tid < 128) {
        const int model = tid >> 6, r = tid & 63;
        float* p = stats + ((((size_t)model * BT_) + row0 + r) * NCHUNKS_ + chunk) * 2;
        p[0] = rm;
        p[1] = rs;
    }
}

// ---------------- Stage 2: fold chunk partials into per-row logZ -----------
__global__ void reduce_logz_kernel(const float* __restrict__ stats,
                                   float* __restrict__ logZ /* [2][BT] */)
{
    const int idx = blockIdx.x * blockDim.x + threadIdx.x;   // 0..2*BT-1
    if (idx >= 2 * BT_) return;
    const float* p = stats + (size_t)idx * NCHUNKS_ * 2;
    float m = -__builtin_inff(), s = 0.0f;
    for (int c = 0; c < NCHUNKS_; ++c) {
        const float cm = p[2 * c], cs = p[2 * c + 1];
        const float nm = fmaxf(m, cm);
        s = s * __expf(m - nm) + cs * __expf(cm - nm);
        m = nm;
    }
    logZ[idx] = m + __logf(s);
}

// ---------------- Hard CE: one wave per row, dot with target weight row ----
__global__ __launch_bounds__(256)
void hard_ce_kernel(const float* __restrict__ S, const float* __restrict__ Ws,
                    const int* __restrict__ tgt, const float* __restrict__ logZ,
                    float* __restrict__ accum /* [1]=hard_sum, [2]=n_valid */)
{
    const int lane = threadIdx.x & 31;
    const int row  = blockIdx.x * (blockDim.x >> 5) + (threadIdx.x >> 5);
    if (row >= BT_) return;
    const int t = tgt[row];
    if (t == -100) return;   // uniform across the wave (one row per wave)

    const float* a = S  + (size_t)row * KS_;
    const float* w = Ws + (size_t)t   * KS_;
    float d = 0.0f;
    for (int k = lane * 4; k < KS_; k += 32 * 4) {
        const float4 av = *(const float4*)(a + k);
        const float4 wv = *(const float4*)(w + k);
        d += av.x * wv.x + av.y * wv.y + av.z * wv.z + av.w * wv.w;
    }
    #pragma unroll
    for (int off = 16; off > 0; off >>= 1) d += __shfl_down(d, off, 32);
    if (lane == 0) {
        atomicAdd(&accum[1], logZ[row] - d);   // -(logit - logZ)
        atomicAdd(&accum[2], 1.0f);
    }
}

// ---------------- Pass 2: recompute logits tile, accumulate JSD ------------
__global__ __launch_bounds__(256)
void pass2_kernel(const float* __restrict__ S, const float* __restrict__ T,
                  const float* __restrict__ Ws, const float* __restrict__ Wt,
                  const float* __restrict__ logZ /* [2][BT] */,
                  float* __restrict__ accum /* [0]=jsd_sum */)
{
    __shared__ unsigned short sAbuf[2 * STAGE_];
    __shared__ unsigned short sWbuf[2 * STAGE_];
    __shared__ float wsum[8];

    const int tid  = threadIdx.x;
    const int lane = tid & 31;
    const int wv   = tid >> 5;
    const int mi   = (wv & 3) * 16;
    const int hf   = lane >> 4;

    const int row0 = blockIdx.x * BM_;
    const int col0 = blockIdx.y * BN_;

    v8f aS0 = {0.f,0.f,0.f,0.f,0.f,0.f,0.f,0.f}, aS1 = aS0;
    v8f aT0 = aS0, aT1 = aS0;
    gemm_dual(S, Ws, KS_, row0, col0, sAbuf, sWbuf, aS0, aS1);
    gemm_dual(T, Wt, KT_, row0, col0, sAbuf, sWbuf, aT0, aT1);

    const float* lzs = logZ + row0;
    const float* lzt = logZ + BT_ + row0;

    float local = 0.0f;
    #pragma unroll
    for (int i = 0; i < 8; ++i) {
        const int r = mi + i + hf * 8;
        const float zs = lzs[r], zt = lzt[r];
        {   // subtile 0
            const float slp = aS0[i] - zs;
            const float tlp = aT0[i] - zt;
            const float pp = __expf(tlp);
            const float qq = __expf(slp);
            const float lm = __logf(0.5f * pp + 0.5f * qq);   // BETA = 0.5
            local += 0.5f * pp * (tlp - lm) + 0.5f * qq * (slp - lm);
        }
        {   // subtile 1 (cols +16)
            const float slp = aS1[i] - zs;
            const float tlp = aT1[i] - zt;
            const float pp = __expf(tlp);
            const float qq = __expf(slp);
            const float lm = __logf(0.5f * pp + 0.5f * qq);
            local += 0.5f * pp * (tlp - lm) + 0.5f * qq * (slp - lm);
        }
    }
    #pragma unroll
    for (int off = 16; off > 0; off >>= 1) local += __shfl_down(local, off, 32);
    if (lane == 0) wsum[wv] = local;
    __syncthreads();
    if (tid == 0) {
        float s = 0.0f;
        #pragma unroll
        for (int i = 0; i < 8; ++i) s += wsum[i];
        atomicAdd(&accum[0], s);
    }
}

// ---------------- Init / finalize ------------------------------------------
__global__ void init_kernel(float* __restrict__ accum) {
    if (threadIdx.x < 4) accum[threadIdx.x] = 0.0f;
}

__global__ void finalize_kernel(const float* __restrict__ accum,
                                float* __restrict__ out) {
    const float nv = fmaxf(accum[2], 1.0f);
    out[0] = 0.5f * (accum[1] / nv) + 0.5f * (accum[0] / (float)BT_);
}

// ---------------------------------------------------------------------------
extern "C" void kernel_launch(void* const* d_in, const int* in_sizes, int n_in,
                              void* d_out, int out_size, void* d_ws, size_t ws_size,
                              hipStream_t stream) {
    const float* S   = (const float*)d_in[0];   // (4096, 2048)
    const float* T   = (const float*)d_in[1];   // (4096, 4096)
    const int*   tgt = (const int*)  d_in[2];   // (4096,)
    const float* Ws  = (const float*)d_in[3];   // (32000, 2048)
    const float* Wt  = (const float*)d_in[4];   // (32000, 4096)
    float* out = (float*)d_out;

    // Workspace layout (floats): [0..3]=accum, [16..16+8192)=logZ[2][BT],
    // then stats[2][BT][NCHUNKS][2]  (~336k floats total, < 1.4 MB).
    float* ws    = (float*)d_ws;
    float* accum = ws;
    float* logZ  = ws + 16;
    float* stats = ws + 16 + 2 * BT_;

    init_kernel<<<1, 64, 0, stream>>>(accum);

    dim3 g1(BT_ / BM_, NCHUNKS_);                 // 64 x 20
    pass1_kernel<<<g1, 256, 0, stream>>>(S, T, Ws, Wt, stats);

    reduce_logz_kernel<<<(2 * BT_ + 255) / 256, 256, 0, stream>>>(stats, logZ);

    hard_ce_kernel<<<BT_ / 8, 256, 0, stream>>>(S, Ws, tgt, logZ, accum);

    dim3 g2(BT_ / BM_, NTILES_);                  // 64 x 500
    pass2_kernel<<<g2, 256, 0, stream>>>(S, T, Ws, Wt, logZ, accum);

    finalize_kernel<<<1, 1, 0, stream>>>(accum, out);
}
